// SlidingWindowAttention_69097433858494
// MI455X (gfx1250) — compile-verified, matrix operand order checked by reading, not run on hardware
//
#include <hip/hip_runtime.h>

#define B_   64
#define D_   4096
#define H_   32
#define HKV_ 8
#define DH_  128
#define W_   2048
#define G_   4
#define KSPLIT_ 8     // K-split factor for all projection GEMMs

typedef __attribute__((ext_vector_type(16))) __bf16 v16bf;
typedef __attribute__((ext_vector_type(8)))  float  v8f;

// native bf16 convert: should lower to v_cvt_pk_bf16_f32 / v_cvt_bf16_f32 on gfx1250
__device__ __forceinline__ __bf16 f2bf(float f) { return (__bf16)f; }

// non-temporal load for read-once streams (weights, KV cache, partials)
__device__ __forceinline__ float ldnt(const float* p) {
    return __builtin_nontemporal_load(p);
}

// k-index of element e (0..15) for lane-group g (0/1) in the 16-bit 16x32 A/B layout
__device__ __forceinline__ int koff(int e, int g) {
    return e + ((e < 8) ? 0 : 8) + (g << 3);
}

__device__ __forceinline__ v8f vz8() {
    v8f z;
#pragma unroll
    for (int i = 0; i < 8; ++i) z[i] = 0.0f;
    return z;
}

// ---------------------------------------------------------------------------
// Split-K GEMM: partial[s][M,N] = concat_A[M, ks..ks+Kc] @ Wt[N, ks..ks+Kc]^T
// blockIdx.y = K-chunk s; one wave computes a 16x128 strip of one chunk.
// A element k:  k < kSplitCat -> A[row*ldA + k], else A2[row*ldA2 + (k - kSplitCat)]
// ---------------------------------------------------------------------------
__global__ void gemm_bf16_wmma(const float* __restrict__ A,  int ldA,
                               const float* __restrict__ A2, int ldA2, int kSplitCat,
                               const float* __restrict__ Wt,
                               float* __restrict__ P,     // partials [KSPLIT_][M*N]
                               int M, int N, int K) {
    const int wid  = (blockIdx.x * blockDim.x + threadIdx.x) >> 5;
    const int lane = threadIdx.x & 31;
    const int tilesM  = M >> 4;
    const int groupsN = N >> 7;
    if (wid >= tilesM * groupsN) return;
    const int mt = wid % tilesM;
    const int ng = wid / tilesM;
    const int n16  = lane & 15;
    const int gsel = lane >> 4;
    const int row  = mt * 16 + n16;          // A row for this lane (A layout: m = lane%16)

    const int Kc = K / KSPLIT_;
    const int ks = blockIdx.y * Kc;

    v8f acc[8];
#pragma unroll
    for (int nt = 0; nt < 8; ++nt) acc[nt] = vz8();

    for (int kk = ks; kk < ks + Kc; kk += 32) {
        v16bf afr;
#pragma unroll
        for (int e = 0; e < 16; ++e) {
            const int k = kk + koff(e, gsel);
            const float av = (k < kSplitCat) ? A[(size_t)row * ldA + k]
                                             : A2[(size_t)row * ldA2 + (k - kSplitCat)];
            afr[e] = f2bf(av);
        }
#pragma unroll
        for (int nt = 0; nt < 8; ++nt) {
            const int col = ng * 128 + nt * 16 + n16;          // B column = lane%16
            const float* wrow = Wt + (size_t)col * K + kk;     // Wt row `col` is B column
            v16bf bfr;
#pragma unroll
            for (int e = 0; e < 16; ++e) bfr[e] = f2bf(ldnt(wrow + koff(e, gsel)));
            acc[nt] = __builtin_amdgcn_wmma_f32_16x16x32_bf16(
                false, afr, false, bfr, (short)0, acc[nt], false, false);
        }
    }

    float* Pp = P + (size_t)blockIdx.y * ((size_t)M * N);
#pragma unroll
    for (int nt = 0; nt < 8; ++nt) {
#pragma unroll
        for (int i = 0; i < 8; ++i) {
            const int r = mt * 16 + i + 8 * gsel;   // C layout: m = vgpr + 8*(lane/16)
            const int c = ng * 128 + nt * 16 + n16; // C layout: n = lane%16
            Pp[(size_t)r * N + c] = acc[nt][i];
        }
    }
}

// ---------------------------------------------------------------------------
// Deterministic K-split reduction + optional gate epilogue.
// mode 0: Y[idx] = sum_s P[s][idx]
// mode 1: Y[idx] = Xres[idx] + sigmoid(sum + bias[idx%N]) * Ores[idx]
// ---------------------------------------------------------------------------
__global__ void reduce_split(const float* __restrict__ P, size_t MN, int N,
                             const float* __restrict__ bias,
                             float* __restrict__ Y,
                             const float* __restrict__ Xres,
                             const float* __restrict__ Ores, int mode) {
    const size_t idx = (size_t)blockIdx.x * blockDim.x + threadIdx.x;
    if (idx >= MN) return;
    float s = 0.0f;
#pragma unroll
    for (int i = 0; i < KSPLIT_; ++i) s += ldnt(P + (size_t)i * MN + idx);
    if (mode == 0) {
        Y[idx] = s;
    } else {
        const int c = (int)(idx % (size_t)N);
        const float gg = 1.0f / (1.0f + __expf(-(s + bias[c])));
        Y[idx] = Xres[idx] + gg * Ores[idx];
    }
}

// ---------------------------------------------------------------------------
// RoPE (interleaved pairs) applied in-place to q [B,H,DH] and k_new [B,HKV,DH]
// ---------------------------------------------------------------------------
__global__ void rope_kernel(float* __restrict__ qbuf, float* __restrict__ kbuf,
                            const int* __restrict__ posp) {
    const int QP = B_ * H_ * (DH_ / 2);     // 131072 q pairs
    const int KP = B_ * HKV_ * (DH_ / 2);   //  32768 k pairs
    const int idx = blockIdx.x * blockDim.x + threadIdx.x;
    if (idx >= QP + KP) return;
    const int pos = *posp;
    float* base;
    int i;
    if (idx < QP) { base = qbuf + (size_t)idx * 2;        i = idx & 63; }
    else          { const int j = idx - QP;
                    base = kbuf + (size_t)j * 2;          i = j & 63; }
    const float invf = powf(10000.0f, -(float)(2 * i) / (float)DH_);
    const float ang  = (float)pos * invf;
    float s, c;
    sincosf(ang, &s, &c);
    const float x0 = base[0], x1 = base[1];
    base[0] = x0 * c - x1 * s;
    base[1] = x0 * s + x1 * c;
}

// ---------------------------------------------------------------------------
// GQA decode attention, one block per (b, kv-head), 8 waves (256 threads).
// Effective K/V row w: w<W-1 -> cache[w+1] (circular shift), else k_new/v_new.
// K and V are each streamed from HBM exactly once (non-temporal).
// ---------------------------------------------------------------------------
__global__ void attn_kernel(const float* __restrict__ q,
                            const float* __restrict__ kc,
                            const float* __restrict__ vc,
                            const float* __restrict__ kn,
                            const float* __restrict__ vn,
                            float* __restrict__ attn_out) {
    __shared__ float sS[G_ * W_];         // 32 KB: unnormalized probs
    __shared__ float sPart[8 * G_ * DH_]; // 16 KB: per-wave PV partials
    __shared__ float sSum[G_];

    const int bh = blockIdx.x;
    const int b  = bh / HKV_;
    const int h  = bh % HKV_;
    const int wave = threadIdx.x >> 5;
    const int lane = threadIdx.x & 31;
    const int n16  = lane & 15;
    const int gsel = lane >> 4;
    const int wbase = wave * (W_ / 8);    // 256 keys per wave

    const size_t bhOff = ((size_t)b * HKV_ + h);

    // --- q A-fragments (M=16, 4 valid rows), d split into 4 chunks of 32 ---
    v16bf aq[4];
#pragma unroll
    for (int ch = 0; ch < 4; ++ch) {
#pragma unroll
        for (int e = 0; e < 16; ++e) {
            const int d = ch * 32 + koff(e, gsel);
            const float v = (n16 < G_)
                ? q[((size_t)b * H_ + h * G_ + n16) * DH_ + d] : 0.0f;
            aq[ch][e] = f2bf(v);
        }
    }

    // --- scores = q . K^T (this wave's 256 keys) ---
    const float scale = 0.08838834764831845f;   // 1/sqrt(128)
    for (int t = 0; t < 16; ++t) {
        const int w = wbase + t * 16 + n16;
        const float* krow = (w < W_ - 1)
            ? (kc + (bhOff * W_ + (w + 1)) * DH_)
            : (kn + bhOff * DH_);
        if (t + 1 < 16) {                         // prefetch next K tile row
            const int wn = w + 16;
            const float* krn = (wn < W_ - 1)
                ? (kc + (bhOff * W_ + (wn + 1)) * DH_)
                : (kn + bhOff * DH_);
            __builtin_prefetch(krn, 0, 1);
        }
        v8f c = vz8();
#pragma unroll
        for (int ch = 0; ch < 4; ++ch) {
            v16bf bfr;
#pragma unroll
            for (int e = 0; e < 16; ++e)
                bfr[e] = f2bf(ldnt(krow + ch * 32 + koff(e, gsel)));
            c = __builtin_amdgcn_wmma_f32_16x16x32_bf16(
                false, aq[ch], false, bfr, (short)0, c, false, false);
        }
        if (gsel == 0) {
#pragma unroll
            for (int i = 0; i < G_; ++i)          // rows m = 0..3 valid
                sS[i * W_ + w] = c[i] * scale;
        }
    }
    __syncthreads();

    // --- softmax stats per row (waves 0..3), exp stored back unnormalized ---
    if (wave < G_) {
        const int g = wave;
        float mx = -1e30f;
        for (int w = lane; w < W_; w += 32) mx = fmaxf(mx, sS[g * W_ + w]);
#pragma unroll
        for (int off = 16; off; off >>= 1) mx = fmaxf(mx, __shfl_xor(mx, off, 32));
        float s = 0.0f;
        for (int w = lane; w < W_; w += 32) {
            const float e = __expf(sS[g * W_ + w] - mx);
            sS[g * W_ + w] = e;
            s += e;
        }
#pragma unroll
        for (int off = 16; off; off >>= 1) s += __shfl_xor(s, off, 32);
        if (lane == 0) sSum[g] = s;
    }
    __syncthreads();

    // --- PV: acc[g][d] += P[g][w] * V[w][d] over this wave's 256 keys ---
    v8f acc[8];
#pragma unroll
    for (int nt = 0; nt < 8; ++nt) acc[nt] = vz8();

    for (int step = 0; step < 8; ++step) {
        const int w0 = wbase + step * 32;
        v16bf ap;
        const float* rows[16];
#pragma unroll
        for (int e = 0; e < 16; ++e) {
            const int w = w0 + koff(e, gsel);
            ap[e] = (n16 < G_) ? f2bf(sS[n16 * W_ + w]) : f2bf(0.0f);
            rows[e] = (w < W_ - 1)
                ? (vc + (bhOff * W_ + (w + 1)) * DH_)
                : (vn + bhOff * DH_);
        }
#pragma unroll
        for (int nt = 0; nt < 8; ++nt) {
            const int d = nt * 16 + n16;
            v16bf bfr;
#pragma unroll
            for (int e = 0; e < 16; ++e) bfr[e] = f2bf(ldnt(rows[e] + d));
            acc[nt] = __builtin_amdgcn_wmma_f32_16x16x32_bf16(
                false, ap, false, bfr, (short)0, acc[nt], false, false);
        }
    }

    // --- deterministic cross-wave reduction via LDS partials ---
    if (gsel == 0) {
#pragma unroll
        for (int nt = 0; nt < 8; ++nt)
#pragma unroll
            for (int i = 0; i < G_; ++i)
                sPart[wave * (G_ * DH_) + i * DH_ + nt * 16 + n16] = acc[nt][i];
    }
    __syncthreads();

    for (int i = threadIdx.x; i < G_ * DH_; i += 256) {
        float s = 0.0f;
#pragma unroll
        for (int wv = 0; wv < 8; ++wv) s += sPart[wv * (G_ * DH_) + i];
        const int m = i >> 7;       // DH_=128
        const int d = i & (DH_ - 1);
        attn_out[((size_t)b * H_ + h * G_ + m) * DH_ + d] = s / sSum[m];
    }
}

// ---------------------------------------------------------------------------
extern "C" void kernel_launch(void* const* d_in, const int* in_sizes, int n_in,
                              void* d_out, int out_size, void* d_ws, size_t ws_size,
                              hipStream_t stream) {
    (void)in_sizes; (void)n_in; (void)out_size; (void)ws_size;
    const float* x   = (const float*)d_in[0];
    const float* kc  = (const float*)d_in[1];
    const float* vc  = (const float*)d_in[2];
    const float* Wq  = (const float*)d_in[3];
    const float* Wk  = (const float*)d_in[4];
    const float* Wv  = (const float*)d_in[5];
    const float* Wo  = (const float*)d_in[6];
    const float* gw  = (const float*)d_in[7];
    const float* gb  = (const float*)d_in[8];
    const int*   pos = (const int*)d_in[9];

    float* ws    = (float*)d_ws;
    float* q_ws  = ws;                               // 64*4096
    float* kn_ws = q_ws  + (size_t)B_ * D_;          // 64*1024
    float* vn_ws = kn_ws + (size_t)B_ * HKV_ * DH_;  // 64*1024
    float* at_ws = vn_ws + (size_t)B_ * HKV_ * DH_;  // 64*4096
    float* o_ws  = at_ws + (size_t)B_ * D_;          // 64*4096
    float* p_ws  = o_ws  + (size_t)B_ * D_;          // KSPLIT_ * 64*4096 partials (8 MB)
    float* out   = (float*)d_out;

    const dim3 blk(256);
    const size_t MNq = (size_t)B_ * D_;              // 262144
    const size_t MNk = (size_t)B_ * HKV_ * DH_;      //  65536

    // q = x @ Wq^T (split-K partials, then reduce)
    gemm_bf16_wmma<<<dim3(16, KSPLIT_), blk, 0, stream>>>(x, D_, x, D_, D_, Wq, p_ws,
                                                          B_, D_, D_);
    reduce_split<<<(int)(MNq / 256), blk, 0, stream>>>(p_ws, MNq, D_, nullptr, q_ws,
                                                       nullptr, nullptr, 0);
    // k_new = x @ Wk^T
    gemm_bf16_wmma<<<dim3(4, KSPLIT_), blk, 0, stream>>>(x, D_, x, D_, D_, Wk, p_ws,
                                                         B_, HKV_ * DH_, D_);
    reduce_split<<<(int)(MNk / 256), blk, 0, stream>>>(p_ws, MNk, HKV_ * DH_, nullptr,
                                                       kn_ws, nullptr, nullptr, 0);
    // v_new = x @ Wv^T
    gemm_bf16_wmma<<<dim3(4, KSPLIT_), blk, 0, stream>>>(x, D_, x, D_, D_, Wv, p_ws,
                                                         B_, HKV_ * DH_, D_);
    reduce_split<<<(int)(MNk / 256), blk, 0, stream>>>(p_ws, MNk, HKV_ * DH_, nullptr,
                                                       vn_ws, nullptr, nullptr, 0);

    // RoPE on q and k_new at position `pos`
    const int pairs = B_ * H_ * (DH_ / 2) + B_ * HKV_ * (DH_ / 2);
    rope_kernel<<<(pairs + 255) / 256, blk, 0, stream>>>(q_ws, kn_ws, pos);

    // attention over the shifted circular cache (512 blocks, streams 1 GB once)
    attn_kernel<<<B_ * HKV_, blk, 0, stream>>>(q_ws, kc, vc, kn_ws, vn_ws, at_ws);

    // o = attn @ Wo^T
    gemm_bf16_wmma<<<dim3(16, KSPLIT_), blk, 0, stream>>>(at_ws, D_, at_ws, D_, D_,
                                                          Wo, p_ws, B_, D_, D_);
    reduce_split<<<(int)(MNq / 256), blk, 0, stream>>>(p_ws, MNq, D_, nullptr, o_ws,
                                                       nullptr, nullptr, 0);

    // out = x + sigmoid([x, o] @ gate_w^T + gate_b) * o   (concat split at k=4096)
    gemm_bf16_wmma<<<dim3(16, KSPLIT_), blk, 0, stream>>>(x, D_, o_ws, D_, D_, gw,
                                                          p_ws, B_, D_, 2 * D_);
    reduce_split<<<(int)(MNq / 256), blk, 0, stream>>>(p_ws, MNq, D_, gb, out,
                                                       x, o_ws, 1);
}
